// biTLSTM_15848429322321
// MI455X (gfx1250) — compile-verified, matrix operand order checked by reading, not run on hardware
//
#include <hip/hip_runtime.h>
#include <hip/hip_bf16.h>
#include <math.h>

// ---------------------------------------------------------------------------
// Problem dims (compile-time)
// ---------------------------------------------------------------------------
#define BB   256
#define SS   256
#define INN  512
#define EE   128
#define HH   512   // hidden size
#define FCN  256
#define OUTN 64

typedef __attribute__((ext_vector_type(16))) __bf16 v16bf;
typedef __attribute__((ext_vector_type(8)))  __bf16 v8bf;
typedef __attribute__((ext_vector_type(4)))  __bf16 v4bf;
typedef __attribute__((ext_vector_type(2)))  __bf16 v2bf;
typedef __attribute__((ext_vector_type(8)))  float  v8f;
typedef __attribute__((ext_vector_type(4)))  float  v4f;

__device__ __forceinline__ v16bf join8(v8bf lo, v8bf hi) {
  return __builtin_shufflevector(lo, hi, 0,1,2,3,4,5,6,7,8,9,10,11,12,13,14,15);
}

// WMMA 16x16x32 bf16 A-fragment (16-bit A, 16x32):
// lane l: row r = l&15, kh = l>>4
//   elems 0..7  = K[kh*8 .. kh*8+7]
//   elems 8..15 = K[16+kh*8 .. 16+kh*8+7]
// B fragments use the same loader on the (N x K) transposed weight matrix.
__device__ __forceinline__ v16bf load_frag_bf16(const __bf16* __restrict__ base,
                                                int stride, int k0) {
  const int lane = threadIdx.x & 31;
  const int r = lane & 15, kh = lane >> 4;
  const __bf16* p = base + (size_t)r * stride + k0 + kh * 8;
  v8bf lo = *(const v8bf*)p;
  v8bf hi = *(const v8bf*)(p + 16);
  return join8(lo, hi);
}

__device__ __forceinline__ v16bf load_frag_f32cvt(const float* __restrict__ base,
                                                  int stride, int k0) {
  const int lane = threadIdx.x & 31;
  const int r = lane & 15, kh = lane >> 4;
  const float* p = base + (size_t)r * stride + k0 + kh * 8;
  v4f x0 = *(const v4f*)(p);
  v4f x1 = *(const v4f*)(p + 4);
  v4f x2 = *(const v4f*)(p + 16);
  v4f x3 = *(const v4f*)(p + 20);
  v16bf out;
#pragma unroll
  for (int i = 0; i < 4; ++i) { out[i]    = (__bf16)x0[i]; out[4+i]  = (__bf16)x1[i]; }
#pragma unroll
  for (int i = 0; i < 4; ++i) { out[8+i]  = (__bf16)x2[i]; out[12+i] = (__bf16)x3[i]; }
  return out;
}

__device__ __forceinline__ v8f wmma_bf16(v16bf a, v16bf b, v8f c) {
  return __builtin_amdgcn_wmma_f32_16x16x32_bf16(false, a, false, b, (short)0, c,
                                                 false, false);
}

__device__ __forceinline__ float sigmoidf_(float x) {
  return 1.0f / (1.0f + __expf(-x));
}

// Branchless tanh: lowers to v_exp_f32 + rcp + bit ops (no libm divergence).
// exp(-2|x|) underflows to 0 for large |x| -> t -> 1, correct saturation.
__device__ __forceinline__ float tanhf_fast(float x) {
  const float e = __expf(-2.0f * fabsf(x));
  const float t = (1.0f - e) / (1.0f + e);
  return copysignf(t, x);
}

// ---------------------------------------------------------------------------
// Small prep kernels
// ---------------------------------------------------------------------------
__global__ void dg_kernel(const float* __restrict__ delta, float* __restrict__ dg, int n) {
  int i = blockIdx.x * 256 + threadIdx.x;
  if (i < n) dg[i] = 1.0f / logf(delta[i] + 2.7182818284590452f);
}

// out[n*rows + k] = (bf16) in[k*istride + n]   (transpose + convert)
__global__ void transpose_cvt(const float* __restrict__ in, __bf16* __restrict__ out,
                              int rows, int cols, int istride) {
  int idx = blockIdx.x * 256 + threadIdx.x;
  if (idx >= rows * cols) return;
  int k = idx % rows;
  int n = idx / rows;
  out[idx] = (__bf16)in[(size_t)k * istride + n];
}

// Pack per-direction gate weights: G[n][0..127]=W[k][n], G[n][128..639]=U[k][n]
__global__ void pack_gates(const float* __restrict__ W, const float* __restrict__ U,
                           __bf16* __restrict__ G) {
  int idx = blockIdx.x * 256 + threadIdx.x;
  if (idx >= 2048 * 640) return;
  int c = idx % 640;
  int n = idx / 640;
  float v = (c < 128) ? W[(size_t)c * 2048 + n] : U[(size_t)(c - 128) * 2560 + n];
  G[idx] = (__bf16)v;
}

// ---------------------------------------------------------------------------
// Generic bf16 WMMA GEMM:  D(M,N) = epi(A(M,K) @ Bt(N,K)^T + bias)
// EPI: 0=none(bf16 out) 1=relu(bf16 out) 2=sigmoid(f32 out)
// ---------------------------------------------------------------------------
template <int EPI, bool AF32>
__global__ __launch_bounds__(256) void gemm_kernel(const void* __restrict__ Ain,
                                                   const __bf16* __restrict__ Bt,
                                                   const float* __restrict__ bias,
                                                   void* __restrict__ Dout,
                                                   int M, int N, int K) {
  const int wid   = blockIdx.x * 8 + (threadIdx.x >> 5);
  const int nstr  = N >> 6;          // 64-wide N strips per wave
  const int mt    = wid / nstr;
  const int ns    = wid % nstr;
  if (mt * 16 >= M) return;
  const int lane  = threadIdx.x & 31;
  const int laneN = lane & 15, hemi = lane >> 4;

  v8f acc[4] = {v8f{}, v8f{}, v8f{}, v8f{}};

  for (int k0 = 0; k0 < K; k0 += 32) {
    v16bf a;
    if constexpr (AF32)
      a = load_frag_f32cvt((const float*)Ain + (size_t)mt * 16 * K, K, k0);
    else
      a = load_frag_bf16((const __bf16*)Ain + (size_t)mt * 16 * K, K, k0);
#pragma unroll
    for (int t = 0; t < 4; ++t) {
      v16bf b = load_frag_bf16(Bt + (size_t)(ns * 64 + t * 16) * K, K, k0);
      acc[t] = wmma_bf16(a, b, acc[t]);
    }
  }

#pragma unroll
  for (int t = 0; t < 4; ++t) {
    const int n  = ns * 64 + t * 16 + laneN;
    const float bv = bias ? bias[n] : 0.0f;
#pragma unroll
    for (int vg = 0; vg < 8; ++vg) {
      const int m = mt * 16 + vg + 8 * hemi;
      float x = acc[t][vg] + bv;
      if constexpr (EPI == 1) x = fmaxf(x, 0.0f);
      if constexpr (EPI == 2) {
        x = sigmoidf_(x);
        ((float*)Dout)[(size_t)m * N + n] = x;
      } else {
        ((__bf16*)Dout)[(size_t)m * N + n] = (__bf16)x;
      }
    }
  }
}

// ---------------------------------------------------------------------------
// Persistent bidirectional TLSTM scan.
// grid.x = 32: blocks 0..15 forward batch tiles, 16..31 backward batch tiles.
// block = 1024 threads = 32 waves; wave w owns hidden units [w*16, w*16+16).
// Per wave per step: 5 accumulator tiles (i,f,g,o over K=640=[v_t|h]; stm over
// K=512=c) -> 40 VGPRs of accumulators; c-state lives in 8 registers/lane.
// ---------------------------------------------------------------------------
#define ASTR 648   // padded LDS stride for [v_t|h] (bank-conflict aware)
#define CSTR 520   // padded LDS stride for c

__global__ __launch_bounds__(1024) void scan_kernel(
    const __bf16* __restrict__ V,       // (B*S, 128) bf16
    const float*  __restrict__ DG,      // (B, S) f32
    const __bf16* __restrict__ Gf, const __bf16* __restrict__ Gb,   // (2048, 640)
    const __bf16* __restrict__ Sf, const __bf16* __restrict__ Sb,   // (512, 512)
    const float*  __restrict__ bf_, const float* __restrict__ bb_,  // (2560,)
    __bf16* __restrict__ H)             // (B*S, 1024): [hf | hb]
{
  const int dir = blockIdx.x >> 4;
  const int bt  = blockIdx.x & 15;
  const __bf16* G  = dir ? Gb : Gf;
  const __bf16* Sp = dir ? Sb : Sf;
  const float*  bias = dir ? bb_ : bf_;

  __shared__ __align__(16) __bf16 Abuf[16 * ASTR];  // cols 0..127 v_t, 128..639 h
  __shared__ __align__(16) __bf16 Cbuf[16 * CSTR];  // c (bf16)
  __shared__ float dtile[16];

  const int tid   = threadIdx.x;
  const int lane  = tid & 31;
  const int laneN = lane & 15, hemi = lane >> 4;
  // wave-uniform hidden-unit block (force into SGPR)
  const int jb = __builtin_amdgcn_readfirstlane((tid >> 5) * 16);

  for (int i = tid; i < 16 * ASTR; i += 1024) Abuf[i] = (__bf16)0.0f;
  for (int i = tid; i < 16 * CSTR; i += 1024) Cbuf[i] = (__bf16)0.0f;

  // SGPR base pointers for this wave's weight rows
  const __bf16* GI = G + (size_t)(jb) * 640;
  const __bf16* GF = G + (size_t)(512 + jb) * 640;
  const __bf16* GG = G + (size_t)(1024 + jb) * 640;
  const __bf16* GO = G + (size_t)(1536 + jb) * 640;
  const __bf16* SP = Sp + (size_t)(jb) * 512;

  // per-lane biases at j = jb + laneN
  const int j = jb + laneN;
  const float bI = bias[j];
  const float bF = bias[512 + j];
  const float bG = bias[1024 + j];
  const float bO = bias[1536 + j];
  const float bS = bias[2048 + j];

  // v_t staging coordinates for this thread
  const int vm = tid >> 6;         // 0..15 (row of the batch tile)
  const int vc = (tid & 63) * 2;   // 0..126 (column pair)
  const __bf16* vrow = V + (size_t)(bt * 16 + vm) * SS * 128 + vc;

  float cst[8];
#pragma unroll
  for (int vg = 0; vg < 8; ++vg) cst[vg] = 0.0f;

  for (int tt = 0; tt < SS; ++tt) {
    const int t = dir ? (SS - 1 - tt) : tt;

    // --- W1: stage v_t (16x128 bf16) and d_t into LDS ---
    *(v2bf*)&Abuf[vm * ASTR + vc] = *(const v2bf*)(vrow + (size_t)t * 128);
    if (tid < 16) dtile[tid] = DG[(size_t)(bt * 16 + tid) * SS + t];
    // prefetch next step's v_t rows (global_prefetch_b8) to hide W1 latency
    if (tt + 1 < SS) {
      const int tn = dir ? (SS - 2 - tt) : (tt + 1);
      __builtin_prefetch(vrow + (size_t)tn * 128, 0, 3);
    }
    __syncthreads();   // B1: all LDS writes (v_t + prev h/c) visible

    // --- R: WMMA GEMMs ---
    v8f aI = v8f{}, aF = v8f{}, aG = v8f{}, aO = v8f{}, aS = v8f{};

#pragma unroll 1
    for (int kt = 0; kt < 20; ++kt) {          // gates: K = 640
      const int k0 = kt * 32;
      v16bf a = load_frag_bf16(Abuf, ASTR, k0);
      aI = wmma_bf16(a, load_frag_bf16(GI, 640, k0), aI);
      aF = wmma_bf16(a, load_frag_bf16(GF, 640, k0), aF);
      aG = wmma_bf16(a, load_frag_bf16(GG, 640, k0), aG);
      aO = wmma_bf16(a, load_frag_bf16(GO, 640, k0), aO);
    }
#pragma unroll 1
    for (int kt = 0; kt < 16; ++kt) {          // stm: K = 512 over c
      const int k0 = kt * 32;
      v16bf a = load_frag_bf16(Cbuf, CSTR, k0);
      aS = wmma_bf16(a, load_frag_bf16(SP, 512, k0), aS);
    }
    __syncthreads();   // B2: all LDS reads done before state rewrite

    // --- W2: lane-local LSTM update; refresh LDS state; emit h ---
    // (dtile read here is safe: next write to dtile happens after B3)
#pragma unroll
    for (int vg = 0; vg < 8; ++vg) {
      const int m = vg + 8 * hemi;
      const float d  = dtile[m];
      const float iv = sigmoidf_(aI[vg] + bI);
      const float fv = sigmoidf_(aF[vg] + bF);
      const float gv = tanhf_fast(aG[vg] + bG);
      const float ov = sigmoidf_(aO[vg] + bO);
      const float sv = tanhf_fast(aS[vg] + bS);
      float c = cst[vg];
      c = fv * (c - sv + sv * d) + iv * gv;
      cst[vg] = c;
      const float h = ov * tanhf_fast(c);
      const __bf16 hb = (__bf16)h;
      Abuf[m * ASTR + 128 + j] = hb;
      Cbuf[m * CSTR + j] = (__bf16)c;
      H[((size_t)((bt * 16 + m) * SS + t)) * 1024 + dir * 512 + j] = hb;
    }
    __syncthreads();   // B3: state rewritten before next iteration's staging
  }
}

// ---------------------------------------------------------------------------
// Host launcher
// ---------------------------------------------------------------------------
extern "C" void kernel_launch(void* const* d_in, const int* in_sizes, int n_in,
                              void* d_out, int out_size, void* d_ws, size_t ws_size,
                              hipStream_t stream) {
  const float* delta    = (const float*)d_in[0];
  const float* batch_xs = (const float*)d_in[1];
  const float* embed    = (const float*)d_in[2];
  const float* W_f      = (const float*)d_in[3];
  const float* U_f      = (const float*)d_in[4];
  const float* b_f      = (const float*)d_in[5];
  const float* W_b      = (const float*)d_in[6];
  const float* U_b      = (const float*)d_in[7];
  const float* b_b      = (const float*)d_in[8];
  const float* fc_w     = (const float*)d_in[9];
  const float* fc_b     = (const float*)d_in[10];
  const float* out_w    = (const float*)d_in[11];
  const float* out_b    = (const float*)d_in[12];

  const int ROWS = BB * SS;  // 65536

  char* ws = (char*)d_ws;
  size_t off = 0;
  auto alloc = [&](size_t bytes) -> void* {
    void* p = ws + off;
    off += (bytes + 255) & ~(size_t)255;
    return p;
  };
  __bf16* Vb   = (__bf16*)alloc((size_t)ROWS * EE * 2);        // v (bf16)
  __bf16* Hb   = (__bf16*)alloc((size_t)ROWS * 1024 * 2);      // [hf|hb]
  __bf16* FCb  = (__bf16*)alloc((size_t)ROWS * FCN * 2);       // fc activations
  __bf16* Gf   = (__bf16*)alloc((size_t)2048 * 640 * 2);
  __bf16* Gb   = (__bf16*)alloc((size_t)2048 * 640 * 2);
  __bf16* Sf   = (__bf16*)alloc((size_t)512 * 512 * 2);
  __bf16* Sb   = (__bf16*)alloc((size_t)512 * 512 * 2);
  __bf16* embT = (__bf16*)alloc((size_t)EE * INN * 2);         // (128,512)
  __bf16* fcT  = (__bf16*)alloc((size_t)FCN * 1024 * 2);       // (256,1024)
  __bf16* outT = (__bf16*)alloc((size_t)OUTN * FCN * 2);       // (64,256)
  float*  dg   = (float*)alloc((size_t)ROWS * 4);

  // 1) dg = 1/log(delta + e)
  dg_kernel<<<(ROWS + 255) / 256, 256, 0, stream>>>(delta, dg, ROWS);

  // 2) weight repacks (transpose + bf16)
  transpose_cvt<<<(INN * EE + 255) / 256, 256, 0, stream>>>(embed, embT, INN, EE, EE);
  transpose_cvt<<<(1024 * FCN + 255) / 256, 256, 0, stream>>>(fc_w, fcT, 1024, FCN, FCN);
  transpose_cvt<<<(FCN * OUTN + 255) / 256, 256, 0, stream>>>(out_w, outT, FCN, OUTN, OUTN);
  transpose_cvt<<<(512 * 512 + 255) / 256, 256, 0, stream>>>(U_f + 2048, Sf, 512, 512, 2560);
  transpose_cvt<<<(512 * 512 + 255) / 256, 256, 0, stream>>>(U_b + 2048, Sb, 512, 512, 2560);
  pack_gates<<<(2048 * 640 + 255) / 256, 256, 0, stream>>>(W_f, U_f, Gf);
  pack_gates<<<(2048 * 640 + 255) / 256, 256, 0, stream>>>(W_b, U_b, Gb);

  // 3) v = batch_xs @ embed   (M=65536, N=128, K=512), f32 A converted on load
  {
    int waves = (ROWS / 16) * (EE / 64);
    gemm_kernel<0, true><<<waves / 8, 256, 0, stream>>>(batch_xs, embT, nullptr, Vb,
                                                        ROWS, EE, INN);
  }

  // 4) bidirectional TLSTM scan (persistent, 32 workgroups x 32 waves)
  scan_kernel<<<32, 1024, 0, stream>>>(Vb, dg, Gf, Gb, Sf, Sb, b_f, b_b, Hb);

  // 5) fc = relu(H @ fc_w + fc_b)   (M=65536, N=256, K=1024)
  {
    int waves = (ROWS / 16) * (FCN / 64);
    gemm_kernel<1, false><<<waves / 8, 256, 0, stream>>>(Hb, fcT, fc_b, FCb,
                                                         ROWS, FCN, 1024);
  }

  // 6) out = sigmoid(fc @ out_w + out_b)  (M=65536, N=64, K=256) -> f32 d_out
  {
    int waves = (ROWS / 16) * (OUTN / 64);
    gemm_kernel<2, false><<<waves / 8, 256, 0, stream>>>(FCb, outT, out_b, d_out,
                                                         ROWS, OUTN, FCN);
  }
}
